// HeadAttention_8005819039800
// MI455X (gfx1250) — matrix operand order repaired and checked by reading, not compiled, not run on hardware
//
#include <hip/hip_runtime.h>
#include <hip/hip_bf16.h>

#define EMB   2048
#define HEAD  128
#define SEQ   4096
#define BATCH 4

typedef __attribute__((ext_vector_type(16))) __bf16 v16bf;
typedef __attribute__((ext_vector_type(8)))  __bf16 v8bf;
typedef __attribute__((ext_vector_type(8)))  float  v8f;

union BF16x16 { v16bf v; v8bf h[2]; };

static __device__ __forceinline__ unsigned short f2bfbits(float f) {
  __bf16 b = (__bf16)f;
  return __builtin_bit_cast(unsigned short, b);
}

static __device__ __forceinline__ v8f wmma_bf16(v16bf a, v16bf b, v8f c) {
  // D = A(16x32 bf16) * B(32x16 bf16) + C(16x16 f32)
  return __builtin_amdgcn_wmma_f32_16x16x32_bf16(false, a, false, b, (short)0, c,
                                                 false, false);
}

static __device__ __forceinline__ v16bf cvt16(float4 a0, float4 a1, float4 a2,
                                              float4 a3) {
  float af[16];
  *(float4*)&af[0]  = a0;
  *(float4*)&af[4]  = a1;
  *(float4*)&af[8]  = a2;
  *(float4*)&af[12] = a3;
  v16bf a;
  #pragma unroll
  for (int e = 0; e < 16; ++e) a[e] = (__bf16)af[e];
  return a;
}

// ---------------------------------------------------------------------------
// Kernel 0: W [E][H] f32 (row-major) -> Wt [H][E] bf16, via 32x32 LDS tile so
// both the fp32 reads and the bf16 writes are coalesced.
// ---------------------------------------------------------------------------
__global__ __launch_bounds__(256) void wt_transpose_kernel(
    const float* __restrict__ W, __bf16* __restrict__ Wt) {
  __shared__ float t[32][33];                    // +1 pad kills bank conflicts
  const int te = blockIdx.x * 32;                // e-tile base
  const int th = blockIdx.y * 32;                // h-tile base
  const int c  = threadIdx.x & 31;
  const int r0 = threadIdx.x >> 5;               // 8 rows per pass
  #pragma unroll
  for (int i = 0; i < 4; ++i) {
    const int r = r0 + i * 8;
    t[r][c] = W[(size_t)(te + r) * HEAD + th + c];
  }
  __syncthreads();
  #pragma unroll
  for (int i = 0; i < 4; ++i) {
    const int r = r0 + i * 8;
    Wt[(size_t)(th + r) * EMB + te + c] = (__bf16)t[c][r];
  }
}

// ---------------------------------------------------------------------------
// Kernel 1: fused QKV projection, no LDS / no barriers, software-pipelined.
// Block = 256 threads = 8 independent waves sharing one 16-row M-tile of x.
// Each wave owns h-tile [16w,16w+16) of Q, K and V (A-frag reuse x3).
// Stage k+1's raw A (fp32) and B-frags load while stage k's WMMAs execute;
// separate "next" registers also break the WMMA->VALU WAR hazard nops.
// ---------------------------------------------------------------------------
__global__ __launch_bounds__(256) void qkv_kernel(
    const float* __restrict__ x,
    const float* __restrict__ bq, const float* __restrict__ bk,
    const float* __restrict__ bv,
    const __bf16* __restrict__ Wqt, const __bf16* __restrict__ Wkt,
    const __bf16* __restrict__ Wvt,
    float* __restrict__ outK, float* __restrict__ outV,
    __bf16* __restrict__ Qbf, __bf16* __restrict__ Kbf,
    __bf16* __restrict__ Vt) {
  const int tid  = threadIdx.x;
  const int wave = tid >> 5, lane = tid & 31;
  const int n    = lane & 15, half = lane >> 4;   // C/D: col n, rows v+8*half
  const int r0   = blockIdx.x * 16;               // global row base (B*S space)

  v8f cq, ck, cv;
  #pragma unroll
  for (int i = 0; i < 8; ++i) { cq[i] = 0.f; ck[i] = 0.f; cv[i] = 0.f; }

  const int h = 16 * wave + n;                    // this lane's output column
  const float*  xa   = x + (size_t)(r0 + n) * EMB + 16 * half;  // A runs
  const float*  xb   = xa + 8;
  const __bf16* wq   = Wqt + (size_t)h * EMB + 16 * half; // B: k = e + 16*half
  const __bf16* wk   = Wkt + (size_t)h * EMB + 16 * half;
  const __bf16* wv   = Wvt + (size_t)h * EMB + 16 * half;

  // -------- pipeline prologue: stage 0 loads --------
  float4 a0 = *(const float4*)(xa);
  float4 a1 = *(const float4*)(xa + 4);
  float4 a2 = *(const float4*)(xb);
  float4 a3 = *(const float4*)(xb + 4);
  v16bf  fq = *(const v16bf*)(wq);
  v16bf  fk = *(const v16bf*)(wk);
  v16bf  fv = *(const v16bf*)(wv);

  for (int k0 = 0; k0 < EMB; k0 += 32) {
    // issue stage k+1 loads (clamped dummy reload on the last iteration)
    const int kn = (k0 + 32 < EMB) ? (k0 + 32) : 0;
    float4 na0 = *(const float4*)(xa + kn);
    float4 na1 = *(const float4*)(xa + kn + 4);
    float4 na2 = *(const float4*)(xb + kn);
    float4 na3 = *(const float4*)(xb + kn + 4);
    v16bf  nq  = *(const v16bf*)(wq + kn);
    v16bf  nk  = *(const v16bf*)(wk + kn);
    v16bf  nv  = *(const v16bf*)(wv + kn);

    // consume stage k
    v16bf a = cvt16(a0, a1, a2, a3);
    cq = wmma_bf16(a, fq, cq);
    ck = wmma_bf16(a, fk, ck);
    cv = wmma_bf16(a, fv, cv);

    a0 = na0; a1 = na1; a2 = na2; a3 = na3;
    fq = nq; fk = nk; fv = nv;
  }

  const float qb = bq[h], kb = bk[h], vb = bv[h];
  const float qscale = 0.08838834764831845f;      // 1/sqrt(128) folded into Q
  const int   b  = r0 / SEQ;
  const int   s0 = r0 % SEQ;

  unsigned short vbits[8];
  #pragma unroll
  for (int v = 0; v < 8; ++v) {
    const size_t gr = (size_t)(r0 + v + 8 * half);
    float qv = cq[v] + qb;
    Qbf[gr * HEAD + h] = (__bf16)(qv * qscale);
    float kv = ck[v] + kb;
    outK[gr * HEAD + h] = kv;
    Kbf[gr * HEAD + h]  = (__bf16)kv;
    float vv = cv[v] + vb;
    outV[gr * HEAD + h] = vv;
    vbits[v] = f2bfbits(vv);
  }
  // V transposed: Vt[(b*H + h)*S + s]; consecutive v -> consecutive s -> pack
  unsigned* vt32 =
      (unsigned*)(Vt + (size_t)(b * HEAD + h) * SEQ + s0 + 8 * half);
  #pragma unroll
  for (int v = 0; v < 8; v += 2)
    vt32[v >> 1] = (unsigned)vbits[v] | ((unsigned)vbits[v + 1] << 16);
}

// ---------------------------------------------------------------------------
// Kernel 2: flash attention. One wave per 16-query tile (8 independent waves
// per block). Online softmax; per-row stats live per C-VGPR slot (row=v+8*half)
// and are reduced across the 16-lane halves with shfl_xor. P goes through a
// wave-private LDS tile to be re-laid from C-layout to A-layout. V^T B-frags
// for the current chunk are loaded BEFORE the softmax section so their latency
// hides under the VALU work (bf16 WMMA is TRANS-class and co-executes).
// ---------------------------------------------------------------------------
__global__ __launch_bounds__(256) void attn_kernel(
    const __bf16* __restrict__ Qbf, const __bf16* __restrict__ Kbf,
    const __bf16* __restrict__ Vt, float* __restrict__ outA) {
  __shared__ __bf16 Ps[8][16][32];                // 1 KB per wave
  const int tid  = threadIdx.x;
  const int wave = tid >> 5, lane = tid & 31;
  const int n    = lane & 15, half = lane >> 4;
  const int tile = blockIdx.x * 8 + wave;         // 0..1023
  const int b    = tile / (SEQ / 16);
  const int q0   = (tile % (SEQ / 16)) * 16;
  const size_t rowbase = (size_t)b * SEQ;

  // Q A-frags for all of H=128 (4 x K=32)
  BF16x16 aq[4];
  const __bf16* qrow = Qbf + (rowbase + q0 + n) * HEAD;
  #pragma unroll
  for (int i = 0; i < 4; ++i) {
    aq[i].h[0] = *(const v8bf*)(qrow + 32 * i + 16 * half);
    aq[i].h[1] = *(const v8bf*)(qrow + 32 * i + 16 * half + 8);
  }

  v8f o[8];
  float mrow[8], lrow[8];
  #pragma unroll
  for (int f = 0; f < 8; ++f)
    for (int i = 0; i < 8; ++i) o[f][i] = 0.f;
  #pragma unroll
  for (int v = 0; v < 8; ++v) { mrow[v] = -1e30f; lrow[v] = 0.f; }

  const int nchunk = (q0 + 16 + 31) / 32;         // causal: keys <= q0+15
  __bf16* pp = &Ps[wave][0][0];

  for (int c = 0; c < nchunk; ++c) {
    const int k0 = c * 32;
    v8f s0, s1;
    #pragma unroll
    for (int i = 0; i < 8; ++i) { s0[i] = 0.f; s1[i] = 0.f; }

    // scores: A=Q, B=K^T (lane col = key, elems = contiguous h run). L2-hot.
    const __bf16* k0row = Kbf + (rowbase + k0 + n) * HEAD + 16 * half;
    const __bf16* k1row = k0row + (size_t)16 * HEAD;
    #pragma unroll
    for (int i = 0; i < 4; ++i) {
      v16bf bk0 = *(const v16bf*)(k0row + 32 * i);
      v16bf bk1 = *(const v16bf*)(k1row + 32 * i);
      s0 = wmma_bf16(aq[i].v, bk0, s0);
      s1 = wmma_bf16(aq[i].v, bk1, s1);
    }

    // hoist V^T B-frag loads: latency hides under the softmax VALU below
    v16bf bvv[8];
    const __bf16* vbase =
        Vt + ((size_t)(b * HEAD) + n) * SEQ + k0 + 16 * half;
    #pragma unroll
    for (int f = 0; f < 8; ++f)
      bvv[f] = *(const v16bf*)(vbase + (size_t)(16 * f) * SEQ);

    if (c == nchunk - 1) {                        // only diagonal chunk masks
      #pragma unroll
      for (int v = 0; v < 8; ++v) {
        const int q = q0 + v + 8 * half;
        if (k0 + n > q)      s0[v] = -1e30f;
        if (k0 + 16 + n > q) s1[v] = -1e30f;
      }
    }

    #pragma unroll
    for (int v = 0; v < 8; ++v) {
      float t = fmaxf(s0[v], s1[v]);
      #pragma unroll
      for (int off = 1; off < 16; off <<= 1)
        t = fmaxf(t, __shfl_xor(t, off, 32));     // row max within 16-lane half
      const float mnew = fmaxf(mrow[v], t);
      const float sc = __expf(mrow[v] - mnew);
      const float p0 = __expf(s0[v] - mnew);
      const float p1 = __expf(s1[v] - mnew);
      float rs = p0 + p1;
      #pragma unroll
      for (int off = 1; off < 16; off <<= 1)
        rs += __shfl_xor(rs, off, 32);            // row sum
      lrow[v] = lrow[v] * sc + rs;
      mrow[v] = mnew;
      #pragma unroll
      for (int f = 0; f < 8; ++f) o[f][v] *= sc;  // rescale accumulators
      pp[(v + 8 * half) * 32 + n]      = (__bf16)p0;   // C-layout -> LDS tile
      pp[(v + 8 * half) * 32 + 16 + n] = (__bf16)p1;
    }

    // P back as A-frag (two aligned ds_load_b128 per lane)
    BF16x16 ap;
    const __bf16* prow = pp + n * 32;
    ap.h[0] = *(const v8bf*)(prow + 16 * half);
    ap.h[1] = *(const v8bf*)(prow + 16 * half + 8);

    // O += P * V
    #pragma unroll
    for (int f = 0; f < 8; ++f) o[f] = wmma_bf16(ap.v, bvv[f], o[f]);
  }

  float inv[8];
  #pragma unroll
  for (int v = 0; v < 8; ++v) inv[v] = 1.0f / lrow[v];
  #pragma unroll
  for (int v = 0; v < 8; ++v) {
    float* orow = outA + (rowbase + q0 + v + 8 * half) * HEAD + n;
    #pragma unroll
    for (int f = 0; f < 8; ++f) orow[16 * f] = o[f][v] * inv[v];
  }
}

// ---------------------------------------------------------------------------
extern "C" void kernel_launch(void* const* d_in, const int* in_sizes, int n_in,
                              void* d_out, int out_size, void* d_ws,
                              size_t ws_size, hipStream_t stream) {
  (void)in_sizes; (void)n_in; (void)out_size; (void)ws_size;
  const float* x  = (const float*)d_in[0];
  const float* Wq = (const float*)d_in[1];
  const float* bq = (const float*)d_in[2];
  const float* Wk = (const float*)d_in[3];
  const float* bk = (const float*)d_in[4];
  const float* Wv = (const float*)d_in[5];
  const float* bv = (const float*)d_in[6];

  const size_t BSH = (size_t)BATCH * SEQ * HEAD;  // 2,097,152
  float* outA = (float*)d_out;                    // [B,S,H] attn
  float* outK = outA + BSH;                       // [B,S,H] K
  float* outV = outA + 2 * BSH;                   // [B,S,H] V

  // Workspace layout (bf16): 3x Wt (512KB) + Qbf/Kbf/Vt (4MB each) = 13.5 MB
  __bf16* ws  = (__bf16*)d_ws;
  const size_t EH = (size_t)EMB * HEAD;
  __bf16* Wqt = ws;
  __bf16* Wkt = Wqt + EH;
  __bf16* Wvt = Wkt + EH;
  __bf16* Qbf = Wvt + EH;
  __bf16* Kbf = Qbf + BSH;
  __bf16* Vt  = Kbf + BSH;

  dim3 wtGrid(EMB / 32, HEAD / 32);
  wt_transpose_kernel<<<wtGrid, 256, 0, stream>>>(Wq, Wqt);
  wt_transpose_kernel<<<wtGrid, 256, 0, stream>>>(Wk, Wkt);
  wt_transpose_kernel<<<wtGrid, 256, 0, stream>>>(Wv, Wvt);

  qkv_kernel<<<(BATCH * SEQ) / 16, 256, 0, stream>>>(
      x, bq, bk, bv, Wqt, Wkt, Wvt, outK, outV, Qbf, Kbf, Vt);

  attn_kernel<<<(BATCH * SEQ) / (16 * 8), 256, 0, stream>>>(Qbf, Kbf, Vt, outA);
}
